// Qwen3Attention_27178553049096
// MI455X (gfx1250) — compile-verified, matrix operand order checked by reading, not run on hardware
//
#include <hip/hip_runtime.h>

// ---------------------------------------------------------------------------
// Types for CDNA5 WMMA (wave32): V_WMMA_F32_16X16X32_BF16
// ---------------------------------------------------------------------------
typedef __attribute__((ext_vector_type(16))) __bf16        bf16x16;
typedef __attribute__((ext_vector_type(8)))  float         f32x8;
typedef __attribute__((ext_vector_type(4)))  unsigned int  u32x4;
typedef __attribute__((ext_vector_type(4)))  int           i32x4;
typedef __attribute__((ext_vector_type(8)))  int           i32x8;

#define SEQ      2048
#define HID      2048
#define HEAD_DIM 128
#define NH       16
#define NKV      8

// Tensor Data Mover availability (probe-verified builtin on both toolchains,
// but with different arity: 5 args on ROCm 7.2 / clang-22, 6 on clang-23).
#if defined(__has_builtin)
#  if __has_builtin(__builtin_amdgcn_tensor_load_to_lds) && \
      __has_builtin(__builtin_amdgcn_s_wait_tensorcnt)
#    define USE_TDM 1
#  else
#    define USE_TDM 0
#  endif
#else
#  define USE_TDM 0
#endif

// round-to-nearest-even float -> bf16 (stored as u16)
__device__ __forceinline__ unsigned short f2bf(float f) {
  union { float f; unsigned u; } c; c.f = f;
  unsigned u = c.u;
  u += 0x7FFFu + ((u >> 16) & 1u);
  return (unsigned short)(u >> 16);
}

// Load a 16x32 WMMA operand fragment (A or B) from LDS.
// ISA 16-bit operand layout: lane l holds k = koff+{0..7} and koff+16+{0..7},
// where caller folds the (lane&16 ? 8 : 0) half-offset into koff.
// Each run is 8 bf16 = 16 bytes -> two ds_read_b128.
__device__ __forceinline__ bf16x16 ld_frag(const unsigned short* sm, int row,
                                           int stride, int koff) {
  union { u32x4 q[2]; bf16x16 v; } u;
  u.q[0] = *reinterpret_cast<const u32x4*>(sm + row * stride + koff);
  u.q[1] = *reinterpret_cast<const u32x4*>(sm + row * stride + koff + 16);
  return u.v;
}

#if USE_TDM
// Issue a TDM 2D tile load global->LDS (bf16 elements, data_size=1 -> 2B).
// D# group0: [1:0]=count=1, [63:32]=lds_addr, [120:64]=global_addr, [127:126]=type=2
// D# group1: [17:16]=data_size, [79:48]=tensor_dim0, [111:80]=tensor_dim1,
//            [127:112]=tile_dim0, [143:128]=tile_dim1, [207:160]=dim0_stride
__device__ __forceinline__ void tdm_load_2d_bf16(const void* gtile,
                                                 unsigned lds_off,
                                                 unsigned tensor_d0,
                                                 unsigned tensor_d1,
                                                 unsigned tile_d0,
                                                 unsigned tile_d1,
                                                 unsigned stride_d0) {
  unsigned long long ga = (unsigned long long)(uintptr_t)gtile;
  u32x4 g0;
  g0[0] = 1u;                                        // count=1 (valid user D#)
  g0[1] = lds_off;                                   // lds_addr
  g0[2] = (unsigned)(ga & 0xFFFFFFFFu);              // global_addr[31:0]
  g0[3] = (unsigned)((ga >> 32) & 0x1FFFFFFu) | (2u << 30);  // addr[56:32]|type=2
  i32x8 g1;
  g1[0] = (int)(1u << 16);                           // data_size=1 (2 bytes)
  g1[1] = (int)((tensor_d0 & 0xFFFFu) << 16);        // tensor_dim0[15:0] @ bit48
  g1[2] = (int)((tensor_d0 >> 16) | ((tensor_d1 & 0xFFFFu) << 16));
  g1[3] = (int)((tensor_d1 >> 16) | (tile_d0 << 16));// tile_dim0 @ bit112
  g1[4] = (int)(tile_d1 & 0xFFFFu);                  // tile_dim1; tile_dim2=0
  g1[5] = (int)stride_d0;                            // dim0_stride[31:0]
  g1[6] = 0;                                         // stride[47:32], dim1_stride lo
  g1[7] = 0;
  i32x4 z4 = {0, 0, 0, 0};
#if __clang_major__ >= 23
  i32x8 z8 = {0, 0, 0, 0, 0, 0, 0, 0};
  __builtin_amdgcn_tensor_load_to_lds(g0, g1, z4, z4, z8, 0);
#else
  __builtin_amdgcn_tensor_load_to_lds(g0, g1, z4, z4, 0);
#endif
}
#endif  // USE_TDM

// ---------------------------------------------------------------------------
// Kernel: fp32 -> bf16 conversion (grid-stride)
// ---------------------------------------------------------------------------
__global__ void f32_to_bf16_kernel(const float* __restrict__ src,
                                   unsigned short* __restrict__ dst,
                                   long long n) {
  long long i = (long long)blockIdx.x * blockDim.x + threadIdx.x;
  long long stride = (long long)gridDim.x * blockDim.x;
  for (; i < n; i += stride) dst[i] = f2bf(src[i]);
}

// ---------------------------------------------------------------------------
// Kernel: C[MxN] = A[MxK] * B[NxK]^T   (bf16 in, fp32 out, WMMA bf16)
// Block: 128 threads (4 waves). Tile: 128(M) x 64(N), K-step 32.
// Wave w computes rows [32w,32w+32) x 64 cols -> 8 WMMAs per K-step from
// 2 A frags + 4 shared B frags (12 ds_read_b128 per 8 WMMAs).
// TDM path: double-buffered LDS; tile t+1 DMA overlaps tile t WMMAs
// (TDM ops from one wave retire in order -> s_wait_tensorcnt 2 gates tile t).
// ---------------------------------------------------------------------------
__global__ __launch_bounds__(128)
void gemm_nt_bf16(float* __restrict__ C,
                  const unsigned short* __restrict__ A,
                  const unsigned short* __restrict__ B,
                  int M, int N, int K, int ldc) {
  __shared__ unsigned short As[USE_TDM ? 2 : 1][128 * 32];
  __shared__ unsigned short Bs[USE_TDM ? 2 : 1][64 * 32];

  const int tid  = threadIdx.x;
  const int lane = tid & 31;
  const int w    = tid >> 5;            // 0..3
  const int kb8  = (lane & 16) ? 8 : 0;
  const int m0   = blockIdx.y * 128;
  const int n0   = blockIdx.x * 64;

  // per-thread global pointers (manual fill + prefetch addresses)
  const unsigned short* ag = A + (size_t)(m0 + tid) * K;              // 1 row/thread
  const unsigned short* bg = B + (size_t)(n0 + (tid >> 1)) * K + (tid & 1) * 16;

  f32x8 acc[2][4] = {};
  const int nk = K >> 5;  // K / 32 tiles

#if USE_TDM
  if (w == 0) {  // prologue: DMA tile 0 into buffer 0
    tdm_load_2d_bf16(A + (size_t)m0 * K, (unsigned)(uintptr_t)As[0],
                     (unsigned)K, (unsigned)M, 32u, 128u, (unsigned)K);
    tdm_load_2d_bf16(B + (size_t)n0 * K, (unsigned)(uintptr_t)Bs[0],
                     (unsigned)K, (unsigned)N, 32u, 64u, (unsigned)K);
  }
#endif

  for (int t = 0; t < nk; ++t) {
    const int k0 = t << 5;
    __syncthreads();  // buffer (t+1)&1 is no longer being read by any wave
#if USE_TDM
    if (w == 0) {
      if (t + 1 < nk) {  // keep next tile's DMA in flight during compute
        tdm_load_2d_bf16(A + (size_t)m0 * K + (k0 + 32),
                         (unsigned)(uintptr_t)As[(t + 1) & 1],
                         (unsigned)K, (unsigned)M, 32u, 128u, (unsigned)K);
        tdm_load_2d_bf16(B + (size_t)n0 * K + (k0 + 32),
                         (unsigned)(uintptr_t)Bs[(t + 1) & 1],
                         (unsigned)K, (unsigned)N, 32u, 64u, (unsigned)K);
        __builtin_amdgcn_s_wait_tensorcnt(2);  // tile t done; t+1 may remain
      } else {
        __builtin_amdgcn_s_wait_tensorcnt(0);
      }
    }
    if (t + 2 < nk) {  // L2 warm-up ahead of the TDM stream
      __builtin_prefetch(ag + k0 + 64, 0, 1);
      __builtin_prefetch(bg + k0 + 64, 0, 1);
    }
    const unsigned short* Ab = As[t & 1];
    const unsigned short* Bb = Bs[t & 1];
#else
    {
      const u32x4* a4 = reinterpret_cast<const u32x4*>(ag + k0);
      u32x4* ad = reinterpret_cast<u32x4*>(As[0] + tid * 32);
      ad[0] = a4[0]; ad[1] = a4[1]; ad[2] = a4[2]; ad[3] = a4[3];
      const u32x4* b4 = reinterpret_cast<const u32x4*>(bg + k0);
      u32x4* bd = reinterpret_cast<u32x4*>(Bs[0] + (tid >> 1) * 32 + (tid & 1) * 16);
      bd[0] = b4[0]; bd[1] = b4[1];
      if (t + 1 < nk) {
        __builtin_prefetch(ag + k0 + 32, 0, 1);
        __builtin_prefetch(bg + k0 + 32, 0, 1);
      }
    }
    const unsigned short* Ab = As[0];
    const unsigned short* Bb = Bs[0];
#endif
    __syncthreads();  // tile t resident in LDS for all waves

    bf16x16 a0 = ld_frag(Ab, w * 32 + (lane & 15), 32, kb8);
    bf16x16 a1 = ld_frag(Ab, w * 32 + 16 + (lane & 15), 32, kb8);
#pragma unroll
    for (int j = 0; j < 4; ++j) {
      bf16x16 b = ld_frag(Bb, j * 16 + (lane & 15), 32, kb8);
      acc[0][j] = __builtin_amdgcn_wmma_f32_16x16x32_bf16(
          false, a0, false, b, (short)0, acc[0][j], false, false);
      acc[1][j] = __builtin_amdgcn_wmma_f32_16x16x32_bf16(
          false, a1, false, b, (short)0, acc[1][j], false, false);
    }
  }

#pragma unroll
  for (int i = 0; i < 2; ++i) {
    const int mbase = m0 + w * 32 + i * 16 + kb8;  // C: m = r + (lane>=16 ? 8:0)
#pragma unroll
    for (int r = 0; r < 8; ++r) {
#pragma unroll
      for (int j = 0; j < 4; ++j) {
        C[(size_t)(mbase + r) * ldc + n0 + j * 16 + (lane & 15)] = acc[i][j][r];
      }
    }
  }
}

// ---------------------------------------------------------------------------
// Kernel: per-(b,s,head) RMSNorm (+weight) and RoPE; fp32 -> bf16 head-major.
// One wave32 block per row; thread handles dims d = lane + 32j (j=0..3).
// RoPE pair (d, d+64) maps to (j, j+2) within the same thread.
// hidx 0..15 -> Q head, 16..23 -> K head, 24..31 -> V head (copy only).
// ---------------------------------------------------------------------------
__global__ __launch_bounds__(32)
void norm_rope_kernel(const float* __restrict__ qkv,
                      const float* __restrict__ cosb,
                      const float* __restrict__ sinb,
                      const float* __restrict__ qw,
                      const float* __restrict__ kw,
                      unsigned short* __restrict__ qout,
                      unsigned short* __restrict__ kout,
                      unsigned short* __restrict__ vout) {
  const int idx  = blockIdx.x;
  const int hidx = idx & 31;
  const int bs   = idx >> 5;          // b*SEQ + s
  const int s    = bs & (SEQ - 1);
  const int b    = bs >> 11;
  const int lane = threadIdx.x;

  int col;
  unsigned short* dst;
  if (hidx < 16) {
    col = hidx * 128;
    dst = qout + (((size_t)b * NH + hidx) * SEQ + s) * HEAD_DIM;
  } else if (hidx < 24) {
    col = 2048 + (hidx - 16) * 128;
    dst = kout + (((size_t)b * NKV + (hidx - 16)) * SEQ + s) * HEAD_DIM;
  } else {
    col = 3072 + (hidx - 24) * 128;
    dst = vout + (((size_t)b * NKV + (hidx - 24)) * SEQ + s) * HEAD_DIM;
  }

  const float* src = qkv + (size_t)bs * 4096 + col;
  float v[4];
#pragma unroll
  for (int j = 0; j < 4; ++j) v[j] = src[lane + 32 * j];

  if (hidx < 24) {  // q/k: RMSNorm + RoPE
    float ss = v[0] * v[0] + v[1] * v[1] + v[2] * v[2] + v[3] * v[3];
#pragma unroll
    for (int off = 16; off >= 1; off >>= 1) ss += __shfl_xor(ss, off, 32);
    const float inv = rsqrtf(ss * (1.0f / 128.0f) + 1e-6f);
    const float* wgt = (hidx < 16) ? qw : kw;
#pragma unroll
    for (int j = 0; j < 4; ++j) v[j] = v[j] * inv * wgt[lane + 32 * j];

    const float* cb = cosb + (size_t)bs * 128;
    const float* sb = sinb + (size_t)bs * 128;
    float c[4], sn[4];
#pragma unroll
    for (int j = 0; j < 4; ++j) {
      c[j]  = cb[lane + 32 * j];
      sn[j] = sb[lane + 32 * j];
    }
    const float o0 = v[0] * c[0] - v[2] * sn[0];
    const float o1 = v[1] * c[1] - v[3] * sn[1];
    const float o2 = v[2] * c[2] + v[0] * sn[2];
    const float o3 = v[3] * c[3] + v[1] * sn[3];
    v[0] = o0; v[1] = o1; v[2] = o2; v[3] = o3;
  }

#pragma unroll
  for (int j = 0; j < 4; ++j) dst[lane + 32 * j] = f2bf(v[j]);
}

// ---------------------------------------------------------------------------
// Kernel: causal flash attention, bf16 WMMA, fp32 online softmax.
// Grid: (qt=32, h=16, b=2). Block: 128 threads = 4 waves; wave owns 16 Q rows.
// LDS: K tile 2x 64x128 (32KB, TDM double-buffered) + V^T 128x64 (16KB)
//      + P 4x16x64 (8KB). K tile t+1 DMA overlaps tile t compute.
// ---------------------------------------------------------------------------
__global__ __launch_bounds__(128)
void attn_kernel(unsigned short* __restrict__ outb,
                 const unsigned short* __restrict__ qg,
                 const unsigned short* __restrict__ kg,
                 const unsigned short* __restrict__ vg) {
  __shared__ unsigned short Ks[USE_TDM ? 2 : 1][64 * 128];
  __shared__ unsigned short Vts[128 * 64];
  __shared__ unsigned short Ps[4 * 16 * 64];

  const int qt = blockIdx.x, h = blockIdx.y, b = blockIdx.z;
  const int kh = h >> 1;  // GQA: 2 Q heads per KV head
  const int q0 = qt * 64;

  const int tid  = threadIdx.x;
  const int lane = tid & 31;
  const int w    = tid >> 5;
  const int kb8  = (lane & 16) ? 8 : 0;  // operand K-half / C-row-half offset

  const unsigned short* qbase = qg + ((size_t)b * NH  + h)  * SEQ * HEAD_DIM;
  const unsigned short* kbase = kg + ((size_t)b * NKV + kh) * SEQ * HEAD_DIM;
  const unsigned short* vbase = vg + ((size_t)b * NKV + kh) * SEQ * HEAD_DIM;

  // Preload Q fragments once (A operand, reused across all K tiles).
  bf16x16 qf[4];
  {
    const unsigned short* qrow =
        qbase + (size_t)(q0 + w * 16 + (lane & 15)) * HEAD_DIM;
#pragma unroll
    for (int c = 0; c < 4; ++c) {
      union { u32x4 q[2]; bf16x16 v; } u;
      u.q[0] = *reinterpret_cast<const u32x4*>(qrow + c * 32 + kb8);
      u.q[1] = *reinterpret_cast<const u32x4*>(qrow + c * 32 + kb8 + 16);
      qf[c] = u.v;
    }
  }

  f32x8 o[8] = {};
  float rm[8], rs[8];
#pragma unroll
  for (int r = 0; r < 8; ++r) { rm[r] = -3.0e38f; rs[r] = 0.0f; }
  const float scale = 0.08838834764831845f;  // 1/sqrt(128)

#if USE_TDM
  if (w == 0) {  // prologue: DMA K tile 0 into buffer 0
    tdm_load_2d_bf16(kbase, (unsigned)(uintptr_t)Ks[0],
                     (unsigned)HEAD_DIM, (unsigned)SEQ,
                     (unsigned)HEAD_DIM, 64u, (unsigned)HEAD_DIM);
  }
#endif

  for (int kt = 0; kt <= qt; ++kt) {  // causal: tiles with k0 <= q0+63 only
    const int k0 = kt * 64;
    __syncthreads();  // prior reads of Ks[(kt+1)&1] and Vts are done
#if USE_TDM
    if (w == 0) {
      if (kt + 1 <= qt) {  // next K tile DMA in flight while computing tile kt
        tdm_load_2d_bf16(kbase + (size_t)(k0 + 64) * HEAD_DIM,
                         (unsigned)(uintptr_t)Ks[(kt + 1) & 1],
                         (unsigned)HEAD_DIM, (unsigned)SEQ,
                         (unsigned)HEAD_DIM, 64u, (unsigned)HEAD_DIM);
        __builtin_amdgcn_s_wait_tensorcnt(1);  // tile kt complete
      } else {
        __builtin_amdgcn_s_wait_tensorcnt(0);
      }
    }
    const unsigned short* Kb = Ks[kt & 1];
#else
    const unsigned short* Kb = Ks[0];
#endif
    {  // Fill V^T tile (manual transpose; TDM cannot transpose).
      const int kk = tid & 63, part = tid >> 6;
#if !USE_TDM
      const unsigned short* ksrc = kbase + (size_t)(k0 + kk) * HEAD_DIM + part * 64;
      u32x4* kdst = reinterpret_cast<u32x4*>(Ks[0] + kk * 128 + part * 64);
      const u32x4* ks4 = reinterpret_cast<const u32x4*>(ksrc);
#pragma unroll
      for (int i = 0; i < 8; ++i) kdst[i] = ks4[i];
#endif
      const unsigned short* vsrc = vbase + (size_t)(k0 + kk) * HEAD_DIM + part * 64;
      const u32x4* vs4 = reinterpret_cast<const u32x4*>(vsrc);
#pragma unroll
      for (int i = 0; i < 8; ++i) {
        u32x4 t = vs4[i];
        const unsigned short* tp = reinterpret_cast<const unsigned short*>(&t);
#pragma unroll
        for (int e = 0; e < 8; ++e)
          Vts[(part * 64 + i * 8 + e) * 64 + kk] = tp[e];
      }
    }
    __syncthreads();

    // scores = Q * K^T  (16x64 per wave, 16 WMMAs)
    f32x8 sc[4] = {};
#pragma unroll
    for (int c = 0; c < 4; ++c) {
#pragma unroll
      for (int j = 0; j < 4; ++j) {
        bf16x16 bk = ld_frag(Kb, j * 16 + (lane & 15), 128, c * 32 + kb8);
        sc[j] = __builtin_amdgcn_wmma_f32_16x16x32_bf16(
            false, qf[c], false, bk, (short)0, sc[j], false, false);
      }
    }

    // scale + causal mask (only diagonal tile needs masking)
#pragma unroll
    for (int j = 0; j < 4; ++j) {
#pragma unroll
      for (int r = 0; r < 8; ++r) {
        float s = sc[j][r] * scale;
        if (kt == qt) {
          const int kj = k0 + j * 16 + (lane & 15);
          const int qi = q0 + w * 16 + r + kb8;
          if (kj > qi) s = -3.0e38f;
        }
        sc[j][r] = s;
      }
    }

    // online softmax per row (row spread across 16 lanes)
#pragma unroll
    for (int r = 0; r < 8; ++r) {
      float mx = fmaxf(fmaxf(sc[0][r], sc[1][r]), fmaxf(sc[2][r], sc[3][r]));
#pragma unroll
      for (int off = 8; off >= 1; off >>= 1) mx = fmaxf(mx, __shfl_xor(mx, off, 16));
      const float rmn = fmaxf(rm[r], mx);
      const float sf  = __expf(rm[r] - rmn);
      float sum = 0.0f;
#pragma unroll
      for (int j = 0; j < 4; ++j) {
        const float pp = __expf(sc[j][r] - rmn);
        sc[j][r] = pp;
        sum += pp;
      }
#pragma unroll
      for (int off = 8; off >= 1; off >>= 1) sum += __shfl_xor(sum, off, 16);
      rs[r] = rs[r] * sf + sum;
      rm[r] = rmn;
#pragma unroll
      for (int dj = 0; dj < 8; ++dj) o[dj][r] *= sf;
    }

    // Stage P (bf16) to LDS in A-operand-friendly row-major layout.
    unsigned short* pw = Ps + w * 1024;
#pragma unroll
    for (int j = 0; j < 4; ++j) {
#pragma unroll
      for (int r = 0; r < 8; ++r)
        pw[(r + kb8) * 64 + j * 16 + (lane & 15)] = f2bf(sc[j][r]);
    }
    __syncthreads();

    // O += P * V   (16 WMMAs per wave over the 64-wide K chunk)
#pragma unroll
    for (int kc = 0; kc < 64; kc += 32) {
      bf16x16 pa = ld_frag(pw, lane & 15, 64, kc + kb8);
#pragma unroll
      for (int dj = 0; dj < 8; ++dj) {
        bf16x16 vfrag = ld_frag(Vts, dj * 16 + (lane & 15), 64, kc + kb8);
        o[dj] = __builtin_amdgcn_wmma_f32_16x16x32_bf16(
            false, pa, false, vfrag, (short)0, o[dj], false, false);
      }
    }
    __syncthreads();
  }

  // Normalize and write bf16 output (B,S, NH*D layout for the O-projection).
#pragma unroll
  for (int r = 0; r < 8; ++r) {
    const float inv = 1.0f / rs[r];
    const int qi = q0 + w * 16 + r + kb8;
    unsigned short* orow = outb + ((size_t)(b * SEQ + qi)) * HID + h * HEAD_DIM;
#pragma unroll
    for (int dj = 0; dj < 8; ++dj)
      orow[dj * 16 + (lane & 15)] = f2bf(o[dj][r] * inv);
  }
}

// ---------------------------------------------------------------------------
// Host orchestration
// ---------------------------------------------------------------------------
extern "C" void kernel_launch(void* const* d_in, const int* in_sizes, int n_in,
                              void* d_out, int out_size, void* d_ws, size_t ws_size,
                              hipStream_t stream) {
  const float* x    = (const float*)d_in[0];
  const float* cosb = (const float*)d_in[1];
  const float* sinb = (const float*)d_in[2];
  const float* Wq   = (const float*)d_in[3];
  const float* Wk   = (const float*)d_in[4];
  const float* Wv   = (const float*)d_in[5];
  const float* Wo   = (const float*)d_in[6];
  const float* qnw  = (const float*)d_in[7];
  const float* knw  = (const float*)d_in[8];
  float* out = (float*)d_out;

  const size_t M = (size_t)2 * SEQ;  // 4096 rows

  char* p = (char*)d_ws;
  auto alloc = [&](size_t bytes) {
    char* r = p;
    p += (bytes + 255) & ~(size_t)255;
    return r;
  };
  unsigned short* xb    = (unsigned short*)alloc(M * HID * 2);
  unsigned short* wqkvb = (unsigned short*)alloc((size_t)4096 * HID * 2);
  unsigned short* wob   = (unsigned short*)alloc((size_t)HID * HID * 2);
  float*          qkv   = (float*)alloc(M * 4096 * 4);
  unsigned short* qb    = (unsigned short*)alloc((size_t)2 * NH  * SEQ * HEAD_DIM * 2);
  unsigned short* kb    = (unsigned short*)alloc((size_t)2 * NKV * SEQ * HEAD_DIM * 2);
  unsigned short* vb    = (unsigned short*)alloc((size_t)2 * NKV * SEQ * HEAD_DIM * 2);
  unsigned short* attb  = (unsigned short*)alloc(M * HID * 2);

  // 1) fp32 -> bf16 operands (Wq/Wk/Wv fused into one 4096x2048 weight block)
  f32_to_bf16_kernel<<<1024, 256, 0, stream>>>(x, xb, (long long)M * HID);
  f32_to_bf16_kernel<<<1024, 256, 0, stream>>>(Wq, wqkvb, 2048LL * 2048);
  f32_to_bf16_kernel<<<512,  256, 0, stream>>>(Wk, wqkvb + (size_t)2048 * 2048, 1024LL * 2048);
  f32_to_bf16_kernel<<<512,  256, 0, stream>>>(Wv, wqkvb + (size_t)3072 * 2048, 1024LL * 2048);
  f32_to_bf16_kernel<<<1024, 256, 0, stream>>>(Wo, wob, 2048LL * 2048);

  // 2) fused QKV projection: qkv[4096x4096] = xb * wqkvb^T
  gemm_nt_bf16<<<dim3(4096 / 64, 4096 / 128), 128, 0, stream>>>(
      qkv, xb, wqkvb, 4096, 4096, 2048, 4096);

  // 3) RMSNorm + RoPE, reshape to head-major bf16 q/k/v
  norm_rope_kernel<<<2 * SEQ * 32, 32, 0, stream>>>(
      qkv, cosb, sinb, qnw, knw, qb, kb, vb);

  // 4) causal flash attention
  attn_kernel<<<dim3(SEQ / 64, NH, 2), 128, 0, stream>>>(attb, qb, kb, vb);

  // 5) output projection: out[4096x2048] = attb * wob^T
  gemm_nt_bf16<<<dim3(2048 / 64, 4096 / 128), 128, 0, stream>>>(
      out, attb, wob, 4096, 2048, 2048, 2048);
}
